// DeformConv_14568529068723
// MI455X (gfx1250) — matrix operand order repaired and checked by reading, not compiled, not run on hardware
//
#include <hip/hip_runtime.h>

// ---------------------------------------------------------------------------
// DCNv2 (modulated deformable 3x3) + BN(eval) + ReLU.  B=4, C=64->64, 128x128.
// Both convolutions run on the WMMA f16 pipe (f32 accumulate).
// GEMM K-order is TAP-MAJOR: kk = t*64 + c  =>  per 32-wide K-chunk the tap
// t = kc>>1 is a wave-uniform scalar and c = (kc&1)*32 + kl is bit math;
// no integer division in the hot loops, one descriptor load per chunk,
// one packed ds_store_b128 per thread per chunk for the B tile.
// LDS ping-pong double buffering: one s_barrier per K-chunk.
// ---------------------------------------------------------------------------

#define BATCH 4
#define CIN   64
#define COUT  64
#define HH    128
#define WW    128
#define HW    (HH * WW)
#define KTAPS 9
#define KDIM  (CIN * KTAPS)   // 576
#define KC    32              // WMMA K chunk
#define NCHUNK (KDIM / KC)    // 18
#define TN    64              // spatial positions per workgroup (half an image row)
#define BN_EPS 1e-5f

typedef __attribute__((ext_vector_type(16))) _Float16 v16h;
typedef __attribute__((ext_vector_type(8)))  _Float16 v8h;
typedef __attribute__((ext_vector_type(8)))  float    v8f;

static __device__ __forceinline__ int iclamp(int v, int lo, int hi) {
    return v < lo ? lo : (v > hi ? hi : v);
}

static __device__ __forceinline__ v16h cat8(v8h lo, v8h hi) {
    return __builtin_shufflevector(lo, hi, 0,1,2,3,4,5,6,7,8,9,10,11,12,13,14,15);
}

// ---------------------------------------------------------------------------
// Kernel 1: prep. f32 weights -> f16 tap-major GEMM layouts + BN fold.
//   whm[64][576]  kk = t*64 + c      main conv weights
//   who[32][576]  rows 27..31 zero   offset conv weights
//   bnsc[o] = gamma/sqrt(var+eps),  bnsh[o] = beta + (bias-mean)*bnsc
// ---------------------------------------------------------------------------
__global__ void dcn_prep(const float* __restrict__ wgt,
                         const float* __restrict__ w_off,
                         const float* __restrict__ bias,
                         const float* __restrict__ gamma,
                         const float* __restrict__ beta,
                         const float* __restrict__ rmean,
                         const float* __restrict__ rvar,
                         _Float16* __restrict__ whm, _Float16* __restrict__ who,
                         float* __restrict__ bnsc, float* __restrict__ bnsh)
{
    const int i = blockIdx.x * 256 + threadIdx.x;
    if (i < COUT) {
        const float s = gamma[i] * rsqrtf(rvar[i] + BN_EPS);
        bnsc[i] = s;
        bnsh[i] = beta[i] + (bias[i] - rmean[i]) * s;
    }
    if (i < COUT * KDIM) {
        const int o = i / KDIM, r = i - o * KDIM;
        const int t = r >> 6, c = r & 63;
        whm[i] = (_Float16)wgt[(o * CIN + c) * KTAPS + t];
    } else {
        const int j = i - COUT * KDIM;
        if (j < 32 * KDIM) {
            const int m = j / KDIM, r = j - m * KDIM;
            const int t = r >> 6, c = r & 63;
            who[j] = (m < 27) ? (_Float16)w_off[(m * CIN + c) * KTAPS + t]
                              : (_Float16)0.f;
        }
    }
}

// ---------------------------------------------------------------------------
// Kernel 2: offset/mask conv as WMMA GEMM. Raw (pre-sigmoid) [B][27][H][W].
// 8 waves = 2(M) x 4(N). Descriptor per (tap,n): clamped addr | invalid<<31.
// ---------------------------------------------------------------------------
__global__ __launch_bounds__(256) void dcn_offset_wmma(
    const float* __restrict__ x, const _Float16* __restrict__ who,
    const float* __restrict__ b_off, float* __restrict__ off_out)
{
    constexpr int LD = 40;                       // halves; 80B rows, 16B aligned
    __shared__ __align__(16) _Float16 sA[2][32 * LD];
    __shared__ __align__(16) _Float16 sB[2][TN * LD];
    __shared__ int sOd[KTAPS * TN];

    const int tid   = threadIdx.x;
    const int b     = blockIdx.x / (HW / TN);
    const int pbase = (blockIdx.x % (HW / TN)) * TN;
    const int h     = pbase / WW, w0 = pbase % WW;   // whole WG in one row

    for (int i = tid; i < KTAPS * TN; i += 256) {
        const int t = i >> 6, n = i & 63;
        const int yy = h + t / 3 - 1;
        const int xx = w0 + n + t % 3 - 1;
        const bool ok = ((unsigned)yy < (unsigned)HH) & ((unsigned)xx < (unsigned)WW);
        sOd[i] = (iclamp(yy, 0, HH - 1) * WW + iclamp(xx, 0, WW - 1))
               | (ok ? 0 : (int)0x80000000);
    }

    const int lane  = tid & 31;
    const int wv    = tid >> 5;
    const int mtile = wv >> 2;                   // 0..1
    const int ncol  = (wv & 3) * 16 + (lane & 15);
    const int mrow  = mtile * 16 + (lane & 15);
    const int kbA   = (lane < 16) ? 0 : 8;       // A 16-bit 16x32 layout
    const int kbB   = (lane < 16) ? 0 : 8;       // B 32x16 layout
    const int nB    = tid & 63;                  // staging: my column
    const int clB   = tid >> 6;                  // staging: my kl block (0..3)

    v8f acc = {};
    __syncthreads();

    for (int kc = 0; kc < NCHUNK; ++kc) {
        const int kk0 = kc * KC;
        const int buf = kc & 1;
        const int t   = kc >> 1;                 // uniform tap for whole chunk
        // stage A: 32x32 halves, 4 contiguous per thread (b64)
        {
            const int m = tid >> 3, k4 = (tid & 7) * 4;
            *(int2*)&sA[buf][m * LD + k4] =
                *(const int2*)&who[m * KDIM + kk0 + k4];
        }
        // stage B: regular im2col; 8 consecutive channels -> 1 b128 store
        {
            const int cd = sOd[t * TN + nB];
            const int c0 = (kc & 1) * 32 + clB * 8;
            const float* xc = x + (size_t)(b * CIN + c0) * HW + (cd & 0xffff);
            v8h vals;
            #pragma unroll
            for (int j = 0; j < 8; ++j) {
                const float ld = xc[j * HW];         // immediate offsets
                vals[j] = (_Float16)((cd >= 0) ? ld : 0.f);
            }
            *(v8h*)&sB[buf][nB * LD + clB * 8] = vals;
        }
        __syncthreads();

        const v8h aLo = *(const v8h*)&sA[buf][mrow * LD + kbA];
        const v8h aHi = *(const v8h*)&sA[buf][mrow * LD + kbA + 16];
        const v8h bLo = *(const v8h*)&sB[buf][ncol * LD + kbB];
        const v8h bHi = *(const v8h*)&sB[buf][ncol * LD + kbB + 16];
        acc = __builtin_amdgcn_wmma_f32_16x16x32_f16(
            false, cat8(aLo, aHi), false, cat8(bLo, bHi),
            (short)0, acc, false, false);
    }

    const int p = pbase + ncol;
    #pragma unroll
    for (int vi = 0; vi < 8; ++vi) {
        const int oc = mtile * 16 + vi + ((lane < 16) ? 0 : 8);
        if (oc < 27)
            off_out[((size_t)(b * 27 + oc)) * HW + p] = acc[vi] + b_off[oc];
    }
}

// ---------------------------------------------------------------------------
// Kernel 3: fused deformable-sample + WMMA GEMM + folded BN + ReLU.
// 8 waves = 4(M) x 2(N-pair); each wave two 16x16 tiles (shared A fragment).
// Descriptor per (tap,n): 16B = 4 clamped corner addrs (16b) + 4 f16 bilinear
// weights premultiplied by sigmoid mask, OOB corners zeroed.
// ---------------------------------------------------------------------------
__global__ __launch_bounds__(256) void dcn_main_wmma(
    const float* __restrict__ x, const float* __restrict__ off,
    const _Float16* __restrict__ whm,
    const float* __restrict__ bnsc, const float* __restrict__ bnsh,
    float* __restrict__ out)
{
    constexpr int LD = 40;
    __shared__ __align__(16) _Float16 sA[2][COUT * LD];
    __shared__ __align__(16) _Float16 sB[2][TN * LD];
    __shared__ __align__(16) int4 sCd[KTAPS * TN];

    const int tid   = threadIdx.x;
    const int b     = blockIdx.x / (HW / TN);
    const int pbase = (blockIdx.x % (HW / TN)) * TN;
    const int h     = pbase / WW, w0 = pbase % WW;

    // --- packed sampling descriptors: 9 taps x 64 positions ---------------
    // mmcv layout: conv channel 2t -> dy, 2t+1 -> dx, 18+t -> mask logit.
    for (int i = tid; i < KTAPS * TN; i += 256) {
        const int t = i >> 6, n = i & 63;
        const size_t obase = (size_t)b * 27 * HW + (pbase + n);
        const float offY = off[obase + (size_t)(2 * t) * HW];
        const float offX = off[obase + (size_t)(2 * t + 1) * HW];
        const float mraw = off[obase + (size_t)(18 + t) * HW];
        const float mk = 1.f / (1.f + __expf(-mraw));
        const float py = (float)(h + t / 3 - 1) + offY;
        const float px = (float)(w0 + n + t % 3 - 1) + offX;
        const float fy = floorf(py), fx = floorf(px);
        const int y0 = (int)fy, x0 = (int)fx;
        const float dy = py - fy, dx = px - fx;
        const int y0c = iclamp(y0, 0, HH - 1), y1c = iclamp(y0 + 1, 0, HH - 1);
        const int x0c = iclamp(x0, 0, WW - 1), x1c = iclamp(x0 + 1, 0, WW - 1);
        const float vy0 = ((unsigned)y0 < (unsigned)HH) ? 1.f : 0.f;
        const float vy1 = ((unsigned)(y0 + 1) < (unsigned)HH) ? 1.f : 0.f;
        const float vx0 = ((unsigned)x0 < (unsigned)WW) ? 1.f : 0.f;
        const float vx1 = ((unsigned)(x0 + 1) < (unsigned)WW) ? 1.f : 0.f;
        const float w00 = (1.f - dy) * (1.f - dx) * mk * vy0 * vx0;
        const float w01 = (1.f - dy) * dx * mk * vy0 * vx1;
        const float w10 = dy * (1.f - dx) * mk * vy1 * vx0;
        const float w11 = dy * dx * mk * vy1 * vx1;
        int4 cd;
        cd.x = (y0c * WW + x0c) | ((y0c * WW + x1c) << 16);
        cd.y = (y1c * WW + x0c) | ((y1c * WW + x1c) << 16);
        cd.z = (int)__builtin_bit_cast(unsigned short, (_Float16)w00)
             | ((int)__builtin_bit_cast(unsigned short, (_Float16)w01) << 16);
        cd.w = (int)__builtin_bit_cast(unsigned short, (_Float16)w10)
             | ((int)__builtin_bit_cast(unsigned short, (_Float16)w11) << 16);
        sCd[i] = cd;
    }

    const int lane  = tid & 31;
    const int wv    = tid >> 5;
    const int mtile = wv >> 1;                   // 0..3
    const int nb    = (wv & 1) * 32;             // N pair base
    const int mrow  = mtile * 16 + (lane & 15);
    const int kbA   = (lane < 16) ? 0 : 8;
    const int kbB   = (lane < 16) ? 0 : 8;
    const int nS    = tid & 63;                  // staging: my column
    const int clS   = tid >> 6;                  // staging: my kl block (0..3)

    v8f acc0 = {}, acc1 = {};
    __syncthreads();                             // descriptors visible

    for (int kc = 0; kc < NCHUNK; ++kc) {
        const int kk0 = kc * KC;
        const int buf = kc & 1;
        const int t   = kc >> 1;                 // uniform tap for whole chunk
        // stage A: 64x32 halves, 8 contiguous per thread (b128 ld+st)
        {
            const int m = tid >> 2, k8 = (tid & 3) * 8;
            *(int4*)&sA[buf][m * LD + k8] =
                *(const int4*)&whm[m * KDIM + kk0 + k8];
        }
        // stage B: deformable bilinear; 1 descriptor, 8 consecutive channels,
        // packed into a single b128 LDS store.
        {
            const int4 cd = sCd[t * TN + nS];
            const int a00 = cd.x & 0xffff, a01 = (unsigned)cd.x >> 16;
            const int a10 = cd.y & 0xffff, a11 = (unsigned)cd.y >> 16;
            const _Float16 h00 = __builtin_bit_cast(_Float16, (unsigned short)(cd.z & 0xffff));
            const _Float16 h01 = __builtin_bit_cast(_Float16, (unsigned short)((unsigned)cd.z >> 16));
            const _Float16 h10 = __builtin_bit_cast(_Float16, (unsigned short)(cd.w & 0xffff));
            const _Float16 h11 = __builtin_bit_cast(_Float16, (unsigned short)((unsigned)cd.w >> 16));
            const int c0 = (kc & 1) * 32 + clS * 8;
            const float* xc = x + (size_t)(b * CIN + c0) * HW;
            v8h vals;
            #pragma unroll
            for (int j = 0; j < 8; ++j) {
                const float x00 = xc[j * HW + a00];   // immediate offsets
                const float x01 = xc[j * HW + a01];
                const float x10 = xc[j * HW + a10];
                const float x11 = xc[j * HW + a11];
                const float v = x00 * (float)h00 + x01 * (float)h01
                              + x10 * (float)h10 + x11 * (float)h11;
                vals[j] = (_Float16)v;
            }
            *(v8h*)&sB[buf][nS * LD + clS * 8] = vals;
        }
        __syncthreads();

        const v8h aLo = *(const v8h*)&sA[buf][mrow * LD + kbA];
        const v8h aHi = *(const v8h*)&sA[buf][mrow * LD + kbA + 16];
        const v16h af = cat8(aLo, aHi);
        const int n0 = nb + (lane & 15);
        const v8h b0l = *(const v8h*)&sB[buf][n0 * LD + kbB];
        const v8h b0h = *(const v8h*)&sB[buf][n0 * LD + kbB + 16];
        const v8h b1l = *(const v8h*)&sB[buf][(n0 + 16) * LD + kbB];
        const v8h b1h = *(const v8h*)&sB[buf][(n0 + 16) * LD + kbB + 16];
        acc0 = __builtin_amdgcn_wmma_f32_16x16x32_f16(
            false, af, false, cat8(b0l, b0h), (short)0, acc0, false, false);
        acc1 = __builtin_amdgcn_wmma_f32_16x16x32_f16(
            false, af, false, cat8(b1l, b1h), (short)0, acc1, false, false);
    }

    // epilogue: folded BN + ReLU, NCHW store.
    // D layout: lane 0-15 -> N=lane, VGPR vi -> M=vi; lane 16-31 -> M=vi+8.
    const int p0 = pbase + nb + (lane & 15);
    #pragma unroll
    for (int vi = 0; vi < 8; ++vi) {
        const int o = mtile * 16 + vi + ((lane < 16) ? 0 : 8);
        const float s = bnsc[o], sh = bnsh[o];
        const size_t ob = (size_t)(b * COUT + o) * HW;
        out[ob + p0]      = fmaxf(acc0[vi] * s + sh, 0.f);
        out[ob + p0 + 16] = fmaxf(acc1[vi] * s + sh, 0.f);
    }
}

// ---------------------------------------------------------------------------
extern "C" void kernel_launch(void* const* d_in, const int* in_sizes, int n_in,
                              void* d_out, int out_size, void* d_ws, size_t ws_size,
                              hipStream_t stream)
{
    const float* x      = (const float*)d_in[0];
    const float* w_off  = (const float*)d_in[1];
    const float* b_off  = (const float*)d_in[2];
    const float* weight = (const float*)d_in[3];
    const float* bias   = (const float*)d_in[4];
    const float* gamma  = (const float*)d_in[5];
    const float* beta   = (const float*)d_in[6];
    const float* rmean  = (const float*)d_in[7];
    const float* rvar   = (const float*)d_in[8];
    float* out = (float*)d_out;

    // workspace layout
    float*    off_ws = (float*)d_ws;                                  // 7,077,888 B
    char*     base   = (char*)d_ws + (size_t)BATCH * 27 * HW * sizeof(float);
    _Float16* whm    = (_Float16*)base;                               // 73,728 B
    _Float16* who    = (_Float16*)(base + (size_t)COUT * KDIM * sizeof(_Float16));
    float*    bnsc   = (float*)(base + (size_t)(COUT + 32) * KDIM * sizeof(_Float16));
    float*    bnsh   = bnsc + COUT;

    const int prep_elems = (COUT + 32) * KDIM;                        // 55,296
    dcn_prep<<<(prep_elems + 255) / 256, 256, 0, stream>>>(
        weight, w_off, bias, gamma, beta, rmean, rvar, whm, who, bnsc, bnsh);
    dcn_offset_wmma<<<BATCH * (HW / TN), 256, 0, stream>>>(x, who, b_off, off_ws);
    dcn_main_wmma<<<BATCH * (HW / TN), 256, 0, stream>>>(
        x, off_ws, whm, bnsc, bnsh, out);
}
